// Layer_16260746182868
// MI455X (gfx1250) — matrix operand order repaired and checked by reading, not compiled, not run on hardware
//
#include <hip/hip_runtime.h>
#include <hip/hip_bf16.h>

typedef __attribute__((ext_vector_type(16))) _Float16 v16h;
typedef __attribute__((ext_vector_type(8)))  _Float16 v8h;
typedef __attribute__((ext_vector_type(8)))  float    v8f;
typedef __attribute__((ext_vector_type(4)))  float    f4;
typedef __attribute__((ext_vector_type(4)))  unsigned int u32x4;
typedef __attribute__((ext_vector_type(8)))  int      i32x8;
typedef __attribute__((ext_vector_type(4)))  int      i32x4;

#define BB 4
#define NTOK 2048
#define DCH 1024
#define DFF 4096
#define LN_EPS 1e-5f

#if __has_builtin(__builtin_amdgcn_tensor_load_to_lds)
#define HAVE_TDM 1
#endif

// ---------------------------------------------------------------------------
// TDM: DMA a 2D tile (rows x 32 f16 elements) from global to LDS.
// LDS layout: row pitch = 40 halves (16 data DWORDs + 4 pad DWORDs per row,
// via D# pad_interval=3 (16 DW), pad_amount=3 (4 DW)) -> conflict-free,
// 16B-aligned fragment reads. D# layout per CDNA5 ISA.
// ---------------------------------------------------------------------------
#ifdef HAVE_TDM
__device__ __forceinline__ void tdm_load_2d_f16(
    unsigned lds_addr, const _Float16* gptr,
    unsigned tile_rows, unsigned row_stride_elems, unsigned tensor_rows)
{
  unsigned long long ga = (unsigned long long)(uintptr_t)gptr;
  u32x4 g0;
  g0[0] = 1u;                                   // count=1, user descriptor
  g0[1] = lds_addr;                             // LDS byte address
  g0[2] = (unsigned)(ga & 0xffffffffu);         // global_addr[31:0]
  g0[3] = (unsigned)((ga >> 32) & 0x1ffffffu)   // global_addr[56:32]
        | (2u << 30);                           // type = 2 (image)

  const unsigned long long t_dim0 = row_stride_elems;  // tensor dim0 (elems)
  const unsigned long long t_dim1 = tensor_rows;       // tensor dim1 (rows)
  const unsigned long long tile0  = 32;                // tile dim0 (K elems)
  const unsigned long long tile1  = tile_rows;         // tile dim1 (rows)
  const unsigned long long strd0  = row_stride_elems;  // dim0 stride (elems)
  const unsigned flags = (1u << 16)   // data_size = 1 (2 bytes)
                       | (1u << 20)   // pad_enable
                       | (3u << 22)   // pad_interval: 16 DWORDs
                       | (3u << 25);  // pad_amount: 4 DWORDs
  unsigned long long q0 = (unsigned long long)flags | (t_dim0 << 48);
  unsigned long long q1 = (t_dim0 >> 16) | (t_dim1 << 16) | (tile0 << 48);
  unsigned long long q2 = tile1 | (strd0 << 32);            // tile_dim2 = 0
  unsigned long long q3 = (strd0 >> 32);                    // dim1 stride = 0
  i32x8 g1;
  g1[0] = (int)(unsigned)(q0 & 0xffffffffu);
  g1[1] = (int)(unsigned)(q0 >> 32);
  g1[2] = (int)(unsigned)(q1 & 0xffffffffu);
  g1[3] = (int)(unsigned)(q1 >> 32);
  g1[4] = (int)(unsigned)(q2 & 0xffffffffu);
  g1[5] = (int)(unsigned)(q2 >> 32);
  g1[6] = (int)(unsigned)(q3 & 0xffffffffu);
  g1[7] = (int)(unsigned)(q3 >> 32);
  i32x4 z4 = {0, 0, 0, 0};
  i32x8 z8 = {0, 0, 0, 0, 0, 0, 0, 0};
  __builtin_amdgcn_tensor_load_to_lds(g0, g1, z4, z4, z8, 0);
}
#endif

// ---------------------------------------------------------------------------
// Block-tiled GEMM via TDM double buffering.
// Block: 256 thr (8 waves), tile 128(M) x 128(N), K step 32.
// Wave (mg,ng) = (wave>>2, wave&3): 64(M) x 32(N), acc[4][2].
// LDS tiles: rows of 32 halves at pitch 40 halves (80 B).
// ---------------------------------------------------------------------------
#define GEMM_BODY(KTOT)                                                       \
  __shared__ __align__(128) _Float16 lA[2][128 * 40];                         \
  __shared__ __align__(128) _Float16 lB[2][128 * 40];                         \
  const int tid  = threadIdx.x;                                               \
  const int lane = tid & 31;                                                  \
  const int wave = tid >> 5;                                                  \
  const int mg = wave >> 2, ng = wave & 3;                                    \
  v8f acc[4][2] = {};                                                         \
  const int m  = lane & 15;                                                   \
  const int kg = lane >> 4;                                                   \
  const int S  = (KTOT) / 32;

// Kernel 3: h = gelu(x1 @ W1^T + b1), all-f16 operands, output f16.
__global__ __launch_bounds__(256) void gemm1_gelu_kernel(
    const _Float16* __restrict__ x1h, const _Float16* __restrict__ W1h,
    const float* __restrict__ bias1, _Float16* __restrict__ h)
{
  const int Mbase = (blockIdx.x >> 5) * 128;    // 4096/128 = 32 N-blocks
  const int Nbase = (blockIdx.x & 31) * 128;
  GEMM_BODY(DCH)

#ifdef HAVE_TDM
  if (wave == 0) {
    tdm_load_2d_f16((unsigned)(uintptr_t)&lA[0][0], x1h + (size_t)Mbase * DCH, 128, DCH, BB * NTOK);
    tdm_load_2d_f16((unsigned)(uintptr_t)&lB[0][0], W1h + (size_t)Nbase * DCH, 128, DCH, DFF);
  }
#endif
  for (int s = 0; s < S; ++s) {
    const int kb = s * 32;
    const int cur = s & 1;
#ifdef HAVE_TDM
    if (wave == 0) __builtin_amdgcn_s_wait_tensorcnt(0);
    __syncthreads();
    if (s + 1 < S && wave == 0) {
      tdm_load_2d_f16((unsigned)(uintptr_t)&lA[cur ^ 1][0], x1h + (size_t)Mbase * DCH + kb + 32, 128, DCH, BB * NTOK);
      tdm_load_2d_f16((unsigned)(uintptr_t)&lB[cur ^ 1][0], W1h + (size_t)Nbase * DCH + kb + 32, 128, DCH, DFF);
    }
#else
    __syncthreads();
    {
      int r = tid >> 1, c = (tid & 1) * 16;
      *(v8h*)&lA[cur][r * 40 + c]     = *(const v8h*)(x1h + (size_t)(Mbase + r) * DCH + kb + c);
      *(v8h*)&lA[cur][r * 40 + c + 8] = *(const v8h*)(x1h + (size_t)(Mbase + r) * DCH + kb + c + 8);
      *(v8h*)&lB[cur][r * 40 + c]     = *(const v8h*)(W1h + (size_t)(Nbase + r) * DCH + kb + c);
      *(v8h*)&lB[cur][r * 40 + c + 8] = *(const v8h*)(W1h + (size_t)(Nbase + r) * DCH + kb + c + 8);
    }
    __syncthreads();
#endif
    union { v16h v; v8h h2[2]; } af[4], bf[2];
    #pragma unroll
    for (int mt = 0; mt < 4; ++mt) {
      int r = mg * 64 + mt * 16 + m;
      af[mt].h2[0] = *(const v8h*)(&lA[cur][r * 40 + kg * 8]);
      af[mt].h2[1] = *(const v8h*)(&lA[cur][r * 40 + 16 + kg * 8]);
    }
    #pragma unroll
    for (int nt = 0; nt < 2; ++nt) {
      int r = ng * 32 + nt * 16 + m;
      bf[nt].h2[0] = *(const v8h*)(&lB[cur][r * 40 + kg * 8]);
      bf[nt].h2[1] = *(const v8h*)(&lB[cur][r * 40 + 16 + kg * 8]);
    }
    #pragma unroll
    for (int mt = 0; mt < 4; ++mt)
      #pragma unroll
      for (int nt = 0; nt < 2; ++nt)
        acc[mt][nt] = __builtin_amdgcn_wmma_f32_16x16x32_f16(
            false, af[mt].v, false, bf[nt].v, (short)0, acc[mt][nt], false, false);
  }

  const int tbase = Mbase + mg * 64;
  const int fbase = Nbase + ng * 32;
  #pragma unroll
  for (int nt = 0; nt < 2; ++nt) {
    const int f = fbase + nt * 16 + (lane & 15);
    const float bv = bias1[f];
    #pragma unroll
    for (int mt = 0; mt < 4; ++mt) {
      #pragma unroll
      for (int j = 0; j < 8; ++j) {
        int t = tbase + mt * 16 + (lane >> 4) * 8 + j;
        float vv = acc[mt][nt][j] + bv;
        float g  = 0.5f * vv * (1.0f + erff(vv * 0.70710678118654752f));
        h[(size_t)t * DFF + f] = (_Float16)g;
      }
    }
  }
}

// Kernel 4: mlp = h @ W2^T + b2, all-f16 operands, f32 out.
__global__ __launch_bounds__(256) void gemm2_kernel(
    const _Float16* __restrict__ hsrc, const _Float16* __restrict__ W2h,
    const float* __restrict__ bias2, float* __restrict__ mlp)
{
  const int Mbase = (blockIdx.x >> 3) * 128;    // 1024/128 = 8 N-blocks
  const int Nbase = (blockIdx.x & 7) * 128;
  GEMM_BODY(DFF)

#ifdef HAVE_TDM
  if (wave == 0) {
    tdm_load_2d_f16((unsigned)(uintptr_t)&lA[0][0], hsrc + (size_t)Mbase * DFF, 128, DFF, BB * NTOK);
    tdm_load_2d_f16((unsigned)(uintptr_t)&lB[0][0], W2h + (size_t)Nbase * DFF, 128, DFF, DCH);
  }
#endif
  for (int s = 0; s < S; ++s) {
    const int kb = s * 32;
    const int cur = s & 1;
#ifdef HAVE_TDM
    if (wave == 0) __builtin_amdgcn_s_wait_tensorcnt(0);
    __syncthreads();
    if (s + 1 < S && wave == 0) {
      tdm_load_2d_f16((unsigned)(uintptr_t)&lA[cur ^ 1][0], hsrc + (size_t)Mbase * DFF + kb + 32, 128, DFF, BB * NTOK);
      tdm_load_2d_f16((unsigned)(uintptr_t)&lB[cur ^ 1][0], W2h + (size_t)Nbase * DFF + kb + 32, 128, DFF, DCH);
    }
#else
    __syncthreads();
    {
      int r = tid >> 1, c = (tid & 1) * 16;
      *(v8h*)&lA[cur][r * 40 + c]     = *(const v8h*)(hsrc + (size_t)(Mbase + r) * DFF + kb + c);
      *(v8h*)&lA[cur][r * 40 + c + 8] = *(const v8h*)(hsrc + (size_t)(Mbase + r) * DFF + kb + c + 8);
      *(v8h*)&lB[cur][r * 40 + c]     = *(const v8h*)(W2h + (size_t)(Nbase + r) * DFF + kb + c);
      *(v8h*)&lB[cur][r * 40 + c + 8] = *(const v8h*)(W2h + (size_t)(Nbase + r) * DFF + kb + c + 8);
    }
    __syncthreads();
#endif
    union { v16h v; v8h h2[2]; } af[4], bf[2];
    #pragma unroll
    for (int mt = 0; mt < 4; ++mt) {
      int r = mg * 64 + mt * 16 + m;
      af[mt].h2[0] = *(const v8h*)(&lA[cur][r * 40 + kg * 8]);
      af[mt].h2[1] = *(const v8h*)(&lA[cur][r * 40 + 16 + kg * 8]);
    }
    #pragma unroll
    for (int nt = 0; nt < 2; ++nt) {
      int r = ng * 32 + nt * 16 + m;
      bf[nt].h2[0] = *(const v8h*)(&lB[cur][r * 40 + kg * 8]);
      bf[nt].h2[1] = *(const v8h*)(&lB[cur][r * 40 + 16 + kg * 8]);
    }
    #pragma unroll
    for (int mt = 0; mt < 4; ++mt)
      #pragma unroll
      for (int nt = 0; nt < 2; ++nt)
        acc[mt][nt] = __builtin_amdgcn_wmma_f32_16x16x32_f16(
            false, af[mt].v, false, bf[nt].v, (short)0, acc[mt][nt], false, false);
  }

  const int tbase = Mbase + mg * 64;
  const int dbase = Nbase + ng * 32;
  #pragma unroll
  for (int nt = 0; nt < 2; ++nt) {
    const int d = dbase + nt * 16 + (lane & 15);
    const float bv = bias2[d];
    #pragma unroll
    for (int mt = 0; mt < 4; ++mt) {
      #pragma unroll
      for (int j = 0; j < 8; ++j) {
        int t = tbase + mt * 16 + (lane >> 4) * 8 + j;
        mlp[(size_t)t * DCH + d] = acc[mt][nt][j] + bv;
      }
    }
  }
}

// ---------------------------------------------------------------------------
// Kernel 1: causal Toeplitz token-mixing conv via WMMA + residual scale.
//   y[b,n,d] = x[b,n,d] + scale[n] * sum_{k<=n} x[b,k,d] * w[n-k]
// ---------------------------------------------------------------------------
__global__ __launch_bounds__(128) void conv_res_kernel(
    const float* __restrict__ x, const float* __restrict__ w,
    const float* __restrict__ scale, float* __restrict__ y)
{
  __shared__ _Float16 wldh[NTOK];      // filter as f16, 4 KB
  __shared__ _Float16 xt[128][40];     // x tile transposed, 10 KB

  const int tid  = threadIdx.x;
  const int lane = tid & 31;
  const int wave = tid >> 5;
  const int dbase = blockIdx.x * 128;
  const int nbase = blockIdx.y * 16;
  const int b     = blockIdx.z;
  const float* xb = x + (size_t)b * NTOK * DCH;

  for (int i = tid; i < NTOK; i += 128) wldh[i] = (_Float16)w[i];

  v8f acc0 = {}, acc1 = {};
  const int m   = lane & 15;
  const int kg  = lane >> 4;
  const int k16 = kg * 16;
  const int chan0 = wave * 32 + (lane & 15);
  const int chan1 = chan0 + 16;
  const int ksteps = (nbase + 47) >> 5;

  for (int s = 0; s < ksteps; ++s) {
    const int kb = s << 5;
    __syncthreads();
    #pragma unroll
    for (int i = 0; i < 32; ++i) {
      int lin = i * 128 + tid;
      int kk  = lin >> 7;
      int dd  = lin & 127;
      xt[dd][kk] = (_Float16)xb[(size_t)(kb + kk) * DCH + dbase + dd];
    }
    __syncthreads();

    v16h afr;
    #pragma unroll
    for (int j = 0; j < 16; ++j) {
      int kj  = kg * 16 + j;
      int idx = nbase + m - (kb + kj);
      int idxc = idx >= 0 ? idx : 0;
      _Float16 av = wldh[idxc];
      afr[j] = idx >= 0 ? av : (_Float16)0.0f;
    }
    union { v16h v; v8h h2[2]; } b0u, b1u;
    b0u.h2[0] = *(const v8h*)&xt[chan0][k16];
    b0u.h2[1] = *(const v8h*)&xt[chan0][k16 + 8];
    b1u.h2[0] = *(const v8h*)&xt[chan1][k16];
    b1u.h2[1] = *(const v8h*)&xt[chan1][k16 + 8];

    acc0 = __builtin_amdgcn_wmma_f32_16x16x32_f16(false, afr, false, b0u.v,
                                                  (short)0, acc0, false, false);
    acc1 = __builtin_amdgcn_wmma_f32_16x16x32_f16(false, afr, false, b1u.v,
                                                  (short)0, acc1, false, false);
  }

  const int mofs = (lane >> 4) * 8;
  #pragma unroll
  for (int half = 0; half < 2; ++half) {
    const int ch = dbase + wave * 32 + half * 16 + (lane & 15);
    #pragma unroll
    for (int j = 0; j < 8; ++j) {
      int tok = nbase + mofs + j;
      size_t o = (size_t)b * NTOK * DCH + (size_t)tok * DCH + ch;
      float a = half ? acc1[j] : acc0[j];
      y[o] = x[o] + a * scale[tok];
    }
  }
}

// ---------------------------------------------------------------------------
__global__ __launch_bounds__(256) void cvt_f16_kernel(
    const float* __restrict__ src, _Float16* __restrict__ dst, int n)
{
  int i = (blockIdx.x * 256 + threadIdx.x) * 8;
  if (i < n) {
    f4 a = *(const f4*)(src + i);
    f4 b = *(const f4*)(src + i + 4);
    v8h o;
    #pragma unroll
    for (int j = 0; j < 4; ++j) { o[j] = (_Float16)a[j]; o[4 + j] = (_Float16)b[j]; }
    *(v8h*)(dst + i) = o;
  }
}

// ---------------------------------------------------------------------------
__global__ __launch_bounds__(256) void ln_kernel(
    const float* __restrict__ a, const float* __restrict__ bm,
    const float* __restrict__ alpha_p, int use_b,
    const float* __restrict__ gw, const float* __restrict__ gb,
    float* __restrict__ out, _Float16* __restrict__ out16)
{
  const int row = blockIdx.x;
  const int tid = threadIdx.x;
  const float alpha = use_b ? alpha_p[0] : 0.0f;
  const size_t base = (size_t)row * DCH;

  float v[4];
  float s = 0.0f, q = 0.0f;
  #pragma unroll
  for (int i = 0; i < 4; ++i) {
    int d = tid + i * 256;
    float t = a[base + d];
    if (use_b) t += alpha * bm[base + d];
    v[i] = t;
    s += t;
    q += t * t;
  }
  #pragma unroll
  for (int o = 16; o > 0; o >>= 1) {
    s += __shfl_down(s, o, 32);
    q += __shfl_down(q, o, 32);
  }
  __shared__ float rs[8], rq[8];
  if ((tid & 31) == 0) { rs[tid >> 5] = s; rq[tid >> 5] = q; }
  __syncthreads();
  float S = 0.0f, Q = 0.0f;
  #pragma unroll
  for (int i = 0; i < 8; ++i) { S += rs[i]; Q += rq[i]; }

  const float mu  = S * (1.0f / DCH);
  const float var = Q * (1.0f / DCH) - mu * mu;
  const float inv = rsqrtf(var + LN_EPS);
  #pragma unroll
  for (int i = 0; i < 4; ++i) {
    int d = tid + i * 256;
    float r = (v[i] - mu) * inv * gw[d] + gb[d];
    out[base + d] = r;
    if (out16) out16[base + d] = (_Float16)r;
  }
}

// ---------------------------------------------------------------------------
extern "C" void kernel_launch(void* const* d_in, const int* in_sizes, int n_in,
                              void* d_out, int out_size, void* d_ws, size_t ws_size,
                              hipStream_t stream) {
  (void)in_sizes; (void)n_in; (void)out_size; (void)ws_size;

  const float* x      = (const float*)d_in[0];
  const float* w      = (const float*)d_in[1];
  const float* scale  = (const float*)d_in[2];
  const float* ln1_w  = (const float*)d_in[3];
  const float* ln1_b  = (const float*)d_in[4];
  const float* W1     = (const float*)d_in[5];
  const float* b1     = (const float*)d_in[6];
  const float* W2     = (const float*)d_in[7];
  const float* b2     = (const float*)d_in[8];
  const float* scalar = (const float*)d_in[9];
  const float* ln2_w  = (const float*)d_in[10];
  const float* ln2_b  = (const float*)d_in[11];
  float* out = (float*)d_out;

  const size_t tok = (size_t)BB * NTOK;       // 8192 rows
  float*     y    = (float*)d_ws;             // tok*DCH f32 (aliased as mlp)
  float*     x1   = y + tok * DCH;            // tok*DCH f32
  _Float16*  h    = (_Float16*)(x1 + tok * DCH);   // tok*DFF f16
  _Float16*  x1h  = h + tok * DFF;            // tok*DCH f16
  _Float16*  W1h  = x1h + tok * DCH;          // DFF*DCH f16
  _Float16*  W2h  = W1h + (size_t)DFF * DCH;  // DCH*DFF f16
  float*     mlp  = y;

  conv_res_kernel<<<dim3(DCH / 128, NTOK / 16, BB), 128, 0, stream>>>(x, w, scale, y);
  ln_kernel<<<dim3((unsigned)tok), 256, 0, stream>>>(y, y, scalar, 0, ln1_w, ln1_b, x1, x1h);
  cvt_f16_kernel<<<dim3(DFF * DCH / 8 / 256), 256, 0, stream>>>(W1, W1h, DFF * DCH);
  cvt_f16_kernel<<<dim3(DFF * DCH / 8 / 256), 256, 0, stream>>>(W2, W2h, DFF * DCH);
  gemm1_gelu_kernel<<<dim3((tok / 128) * (DFF / 128)), 256, 0, stream>>>(x1h, W1h, b1, h);
  gemm2_kernel<<<dim3((tok / 128) * (DCH / 128)), 256, 0, stream>>>(h, W2h, b2, mlp);
  ln_kernel<<<dim3((unsigned)tok), 256, 0, stream>>>(x1, mlp, scalar, 1, ln2_w, ln2_b, out, nullptr);
}
